// finalModel_3487513444609
// MI455X (gfx1250) — compile-verified, matrix operand order checked by reading, not run on hardware
//
#include <hip/hip_runtime.h>
#include <math.h>

typedef __attribute__((ext_vector_type(16))) __bf16 v16bf;
typedef __attribute__((ext_vector_type(8)))  float  v8f;

#define TM 64
#define TN 128
#define TK 32
#define APITCH 40                        // LDS row pitch in elements (80 B)
#define AS_BYTES (TM * APITCH * 2)       // 5120
#define BS_BYTES (TN * APITCH * 2)       // 10240
#define STAGE_BYTES (AS_BYTES + BS_BYTES) // 15360 per pipeline stage

__device__ __forceinline__ __bf16 f2bf(float x) {
  union { float f; unsigned u; } v; v.f = x;
  unsigned r = (v.u + 0x7FFFu + ((v.u >> 16) & 1u)) >> 16;
  union { unsigned short s; __bf16 b; } o; o.s = (unsigned short)r;
  return o.b;
}

// async global->LDS copy, 16B per lane; tracked by ASYNCcnt
__device__ __forceinline__ void async_ld_b128(unsigned lds_byte_off, const void* g) {
  asm volatile("global_load_async_to_lds_b128 %0, %1, off"
               :: "v"(lds_byte_off), "v"((unsigned long long)(uintptr_t)g)
               : "memory");
}
__device__ __forceinline__ void wait_async0() {
  asm volatile("s_wait_asynccnt 0" ::: "memory");
}
__device__ __forceinline__ void wait_async3() {   // allow next tile's 3 ops in flight
  asm volatile("s_wait_asynccnt 3" ::: "memory");
}

__device__ __forceinline__ float wred_sum(float v) {
#pragma unroll
  for (int o = 16; o; o >>= 1) v += __shfl_xor(v, o, 32);
  return v;
}
__device__ __forceinline__ float wred_max(float v) {
#pragma unroll
  for (int o = 16; o; o >>= 1) v = fmaxf(v, __shfl_xor(v, o, 32));
  return v;
}
__device__ __forceinline__ int lb_i32(const int* __restrict__ arr, int n, int key) {
  int lo = 0, hi = n;
  while (lo < hi) { int mid = (lo + hi) >> 1; if (arr[mid] < key) lo = mid + 1; else hi = mid; }
  return lo;
}
__device__ __forceinline__ float fsigmoid(float x) {
  return 1.f / (1.f + __expf(-x));
}

// ---------------------------------------------------------------------------
__global__ __launch_bounds__(256)
void cast_kernel(const float* __restrict__ in, __bf16* __restrict__ out, long n)
{
  long i = (long)blockIdx.x * 256 + threadIdx.x;
  if (i < n) out[i] = f2bf(in[i]);
}

// transpose+cast: out[n*K+k] = bf16(in[k*N+n]); in is [K][N] f32
__global__ __launch_bounds__(256)
void tcast_kernel(const float* __restrict__ in, __bf16* __restrict__ out, int K, int N)
{
  long i = (long)blockIdx.x * 256 + threadIdx.x;
  if (i >= (long)K * N) return;
  int k = (int)(i / N), n = (int)(i % N);
  out[(long)n * K + k] = f2bf(in[i]);
}

// Wbar_t[n*256+k] = bf16(mean over 3 of W[j][k][n])
__global__ __launch_bounds__(256)
void wbar_t_kernel(const float* __restrict__ W, __bf16* __restrict__ Wt)
{
  int i = blockIdx.x * 256 + threadIdx.x;        // i over 65536
  int k = i >> 8, n = i & 255;
  float v = (W[i] + W[65536 + i] + W[131072 + i]) * (1.f / 3.f);
  Wt[(long)n * 256 + k] = f2bf(v);
}

// ---------------------------------------------------------------------------
// bf16 WMMA GEMM, double-buffered async-LDS pipeline.
// A [Mr,Kd] bf16 row-major; Bt [256,Kd] bf16 pre-transposed (row = out col).
// Output cols fixed at 256 (grid.x == 2). Cf(f32)/Cb(bf16) optional outputs.
// ACT: 0 none, 1 lrelu(0.01), 2 relu
// ---------------------------------------------------------------------------
template<int ACT>
__global__ __launch_bounds__(256)
void gemm_abf16(const __bf16* __restrict__ A, const __bf16* __restrict__ Bt,
                const float* __restrict__ bias, float* __restrict__ Cf,
                __bf16* __restrict__ Cb, int Mr, int Kd, int ldc, int coff,
                float scale)
{
  __shared__ __align__(16) unsigned char smem[2 * STAGE_BYTES];

  const int tid  = threadIdx.x;
  const int lane = tid & 31, wave = tid >> 5;
  const int wm = wave >> 2, wn = wave & 3;
  const int hl = lane >> 4, l16 = lane & 15;
  const long m0 = (long)blockIdx.y * TM;
  const long n0 = (long)blockIdx.x * TN;

  const int srow = tid >> 2;            // 0..63
  const int sseg = (tid & 3) * 8;       // element offset of this 16B segment
  long arow = m0 + srow;                // clamp: always issue, discard at store
  if (arow >= Mr) arow = Mr - 1;
  const __bf16* aP  = A + arow * (long)Kd + sseg;
  const __bf16* bP0 = Bt + (n0 + srow) * (long)Kd + sseg;
  const __bf16* bP1 = Bt + (n0 + srow + 64) * (long)Kd + sseg;
  const unsigned ldsA  = (unsigned)(srow * (APITCH * 2) + sseg * 2);
  const unsigned ldsB0 = (unsigned)(AS_BYTES + srow * (APITCH * 2) + sseg * 2);
  const unsigned ldsB1 = (unsigned)(AS_BYTES + (srow + 64) * (APITCH * 2) + sseg * 2);

  v8f c[2][2] = {};
  const int nt = Kd / TK;

  // prologue: stage tile 0 into buffer 0
  async_ld_b128(ldsA,  aP);
  async_ld_b128(ldsB0, bP0);
  async_ld_b128(ldsB1, bP1);

  for (int t = 0; t < nt; ++t) {
    if (t + 1 < nt) {     // issue next tile into the other stage, keep 3 in flight
      const unsigned sb = (unsigned)((t + 1) & 1) * STAGE_BYTES;
      const int kn = (t + 1) * TK;
      async_ld_b128(sb + ldsA,  aP  + kn);
      async_ld_b128(sb + ldsB0, bP0 + kn);
      async_ld_b128(sb + ldsB1, bP1 + kn);
      wait_async3();
    } else {
      wait_async0();
    }
    __syncthreads();

    const __bf16* As = (const __bf16*)(smem + (unsigned)(t & 1) * STAGE_BYTES);
    const __bf16* Bs = As + AS_BYTES / 2;

    v16bf af[2], bfv[2];
#pragma unroll
    for (int f = 0; f < 2; ++f) {
      const int mr = wm * 32 + f * 16 + l16;
      const int nc = wn * 32 + f * 16 + l16;
#pragma unroll
      for (int j = 0; j < 8; ++j) {
        const int kk = (j >> 2) * 16 + hl * 8 + (j & 3) * 2;
        af[f][2 * j]      = As[mr * APITCH + kk];
        af[f][2 * j + 1]  = As[mr * APITCH + kk + 1];
        bfv[f][2 * j]     = Bs[nc * APITCH + kk];
        bfv[f][2 * j + 1] = Bs[nc * APITCH + kk + 1];
      }
    }
#pragma unroll
    for (int fm = 0; fm < 2; ++fm)
#pragma unroll
      for (int fn = 0; fn < 2; ++fn)
        c[fm][fn] = __builtin_amdgcn_wmma_f32_16x16x32_bf16(
            false, af[fm], false, bfv[fn], (short)0, c[fm][fn], false, false);
    __syncthreads();   // closes reads of this stage before it is re-staged
  }

#pragma unroll
  for (int fm = 0; fm < 2; ++fm) {
#pragma unroll
    for (int fn = 0; fn < 2; ++fn) {
      const long gcol = n0 + wn * 32 + fn * 16 + l16;
      const float bv = bias ? bias[gcol] : 0.f;
#pragma unroll
      for (int r = 0; r < 8; ++r) {
        const long grow = m0 + wm * 32 + fm * 16 + r + 8 * hl;
        if (grow < Mr) {
          float v = c[fm][fn][r] * scale + bv;
          if (ACT == 1)      v = (v > 0.f) ? v : 0.01f * v;
          else if (ACT == 2) v = fmaxf(v, 0.f);
          const long idx = grow * (long)ldc + coff + gcol;
          if (Cf) Cf[idx] = v;
          if (Cb) Cb[idx] = f2bf(v);
        }
      }
    }
  }
}

// ---------------------------------------------------------------------------
// Fused rel-att edge MLP (bf16 operands, double-buffered async staging):
// g[e,m] = sigmoid( sum_d relu((he@W1m)[d]+b1m[d])*W2m[d] + b2m )
// ---------------------------------------------------------------------------
__global__ __launch_bounds__(256)
void rel_att_kernel(const __bf16* __restrict__ he, const __bf16* __restrict__ W1t,
                    const float* __restrict__ b1, const float* __restrict__ W2,
                    const float* __restrict__ b2, float* __restrict__ g, int E)
{
  __shared__ __align__(16) unsigned char smem[2 * STAGE_BYTES + 256];
  float* rowacc = (float*)(smem + 2 * STAGE_BYTES);

  const int m = blockIdx.y;
  const __bf16* Bt = W1t + (long)m * 256 * 256;
  const float* b1p = b1 + m * 256;
  const float* w2p = W2 + m * 256;
  const float  b2s = b2[m];

  const int tid  = threadIdx.x;
  const int lane = tid & 31, wave = tid >> 5;
  const int wm = wave >> 2, wn = wave & 3;
  const int hl = lane >> 4, l16 = lane & 15;
  const long e0 = (long)blockIdx.x * TM;

  const int srow = tid >> 2;
  const int sseg = (tid & 3) * 8;
  long arow = e0 + srow;
  if (arow >= E) arow = E - 1;
  const __bf16* aP = he + arow * 256 + sseg;
  const unsigned ldsA  = (unsigned)(srow * (APITCH * 2) + sseg * 2);
  const unsigned ldsB0 = (unsigned)(AS_BYTES + srow * (APITCH * 2) + sseg * 2);
  const unsigned ldsB1 = (unsigned)(AS_BYTES + (srow + 64) * (APITCH * 2) + sseg * 2);

  if (tid < TM) rowacc[tid] = 0.f;
  __syncthreads();

  for (int pass = 0; pass < 2; ++pass) {
    const int n0 = pass * TN;
    const __bf16* bP0 = Bt + (n0 + srow) * 256L + sseg;
    const __bf16* bP1 = Bt + (n0 + srow + 64) * 256L + sseg;
    v8f c[2][2] = {};

    async_ld_b128(ldsA,  aP);
    async_ld_b128(ldsB0, bP0);
    async_ld_b128(ldsB1, bP1);

    for (int t = 0; t < 8; ++t) {        // 256 / TK
      if (t + 1 < 8) {
        const unsigned sb = (unsigned)((t + 1) & 1) * STAGE_BYTES;
        const int kn = (t + 1) * TK;
        async_ld_b128(sb + ldsA,  aP  + kn);
        async_ld_b128(sb + ldsB0, bP0 + kn);
        async_ld_b128(sb + ldsB1, bP1 + kn);
        wait_async3();
      } else {
        wait_async0();
      }
      __syncthreads();

      const __bf16* As = (const __bf16*)(smem + (unsigned)(t & 1) * STAGE_BYTES);
      const __bf16* Bs = As + AS_BYTES / 2;

      v16bf af[2], bfv[2];
#pragma unroll
      for (int f = 0; f < 2; ++f) {
        const int mr = wm * 32 + f * 16 + l16;
        const int nc = wn * 32 + f * 16 + l16;
#pragma unroll
        for (int j = 0; j < 8; ++j) {
          const int kk = (j >> 2) * 16 + hl * 8 + (j & 3) * 2;
          af[f][2 * j]      = As[mr * APITCH + kk];
          af[f][2 * j + 1]  = As[mr * APITCH + kk + 1];
          bfv[f][2 * j]     = Bs[nc * APITCH + kk];
          bfv[f][2 * j + 1] = Bs[nc * APITCH + kk + 1];
        }
      }
#pragma unroll
      for (int fm = 0; fm < 2; ++fm)
#pragma unroll
        for (int fn = 0; fn < 2; ++fn)
          c[fm][fn] = __builtin_amdgcn_wmma_f32_16x16x32_bf16(
              false, af[fm], false, bfv[fn], (short)0, c[fm][fn], false, false);
      __syncthreads();
    }
    // relu(t+b1)*W2 reduced over d into per-edge accumulators (ds_add_f32)
#pragma unroll
    for (int fm = 0; fm < 2; ++fm) {
#pragma unroll
      for (int fn = 0; fn < 2; ++fn) {
        const int d = n0 + wn * 32 + fn * 16 + l16;
        const float bb = b1p[d];
        const float ww = w2p[d];
#pragma unroll
        for (int r = 0; r < 8; ++r) {
          const int row = wm * 32 + fm * 16 + r + 8 * hl;
          float t = c[fm][fn][r] + bb;
          t = fmaxf(t, 0.f) * ww;
          atomicAdd(&rowacc[row], t);
        }
      }
    }
    __syncthreads();
  }
  if (tid < TM && e0 + tid < E)
    g[(e0 + tid) * 3 + m] = fsigmoid(rowacc[tid] + b2s);
}

// a[e] = dot(hn[src[e]], hn[dst[e]]) on bf16 features — one wave per edge
__global__ __launch_bounds__(256)
void edge_dot_kernel(const __bf16* __restrict__ hn, const int* __restrict__ src,
                     const int* __restrict__ dst, float* __restrict__ a, int E)
{
  const int e = blockIdx.x * 8 + (threadIdx.x >> 5);
  if (e >= E) return;
  const int lane = threadIdx.x & 31;
  const __bf16* ps = hn + (long)src[e] * 256;
  const __bf16* pd = hn + (long)dst[e] * 256;
  float acc = 0.f;
#pragma unroll
  for (int i = 0; i < 8; ++i)
    acc += (float)ps[lane + 32 * i] * (float)pd[lane + 32 * i];
  acc = wred_sum(acc);
  if (lane == 0) a[e] = acc;
}

// per-destination softmax over contiguous (sorted-dst) edge ranges
__global__ __launch_bounds__(256)
void seg_softmax_kernel(const float* __restrict__ a, const float* __restrict__ g,
                        const int* __restrict__ dst, float* __restrict__ alpha,
                        float* __restrict__ beta, int N, int E)
{
  const int n = blockIdx.x * 8 + (threadIdx.x >> 5);
  if (n >= N) return;
  const int lane = threadIdx.x & 31;
  const int lo = lb_i32(dst, E, n);
  const int hi = lb_i32(dst, E, n + 1);
  if (lo >= hi) return;
  {
    float mx = -1e30f;
    for (int e = lo + lane; e < hi; e += 32) mx = fmaxf(mx, a[e]);
    mx = wred_max(mx);
    float s = 0.f;
    for (int e = lo + lane; e < hi; e += 32) s += __expf(a[e] - mx);
    s = wred_sum(s);
    const float inv = 1.f / s;
    for (int e = lo + lane; e < hi; e += 32) alpha[e] = __expf(a[e] - mx) * inv;
  }
  for (int h = 0; h < 3; ++h) {
    float mx = -1e30f;
    for (int e = lo + lane; e < hi; e += 32) mx = fmaxf(mx, g[e * 3 + h]);
    mx = wred_max(mx);
    float s = 0.f;
    for (int e = lo + lane; e < hi; e += 32) s += __expf(g[e * 3 + h] - mx);
    s = wred_sum(s);
    const float inv = 1.f / s;
    for (int e = lo + lane; e < hi; e += 32) beta[e * 3 + h] = __expf(g[e * 3 + h] - mx) * inv;
  }
}

// agg_a[n,:] = sum alpha_e*hn[src[e]]; agg_m[n,m,:] = sum beta[e,m]*hn[src[e]]
__global__ __launch_bounds__(256)
void agg_kernel(const __bf16* __restrict__ hn, const int* __restrict__ src,
                const int* __restrict__ dst, const float* __restrict__ alpha,
                const float* __restrict__ beta, __bf16* __restrict__ agg_a,
                __bf16* __restrict__ agg_m, int E)
{
  const int n = blockIdx.x;
  const int d = threadIdx.x;
  const int lo = lb_i32(dst, E, n);
  const int hi = lb_i32(dst, E, n + 1);
  float aA = 0.f, a0 = 0.f, a1 = 0.f, a2 = 0.f;
  for (int e = lo; e < hi; ++e) {
    const float v = (float)hn[(long)src[e] * 256 + d];
    aA += alpha[e] * v;
    a0 += beta[e * 3 + 0] * v;
    a1 += beta[e * 3 + 1] * v;
    a2 += beta[e * 3 + 2] * v;
  }
  agg_a[(long)n * 256 + d] = f2bf(aA);
  agg_m[(long)n * 768 + d]       = f2bf(a0);
  agg_m[(long)n * 768 + 256 + d] = f2bf(a1);
  agg_m[(long)n * 768 + 512 + d] = f2bf(a2);
}

// h = gate*where(has_in, h_red, hn) + (1-gate)*h ; refresh bf16 copy
__global__ __launch_bounds__(256)
void gate_update_kernel(float* __restrict__ h, __bf16* __restrict__ h_bf,
                        const float* __restrict__ h_red, const float* __restrict__ hn,
                        const int* __restrict__ dst, const float* __restrict__ gW,
                        const float* __restrict__ gb, int E)
{
  const int n = blockIdx.x;
  const int t = threadIdx.x;
  __shared__ float red[256];
  __shared__ float sg;
  __shared__ int has;
  const float hv = h[(long)n * 256 + t];
  red[t] = hv * gW[t];
  __syncthreads();
  for (int s = 128; s; s >>= 1) { if (t < s) red[t] += red[t + s]; __syncthreads(); }
  if (t == 0) {
    sg = fsigmoid(red[0] + gb[0]);
    has = (lb_i32(dst, E, n) != lb_i32(dst, E, n + 1));
  }
  __syncthreads();
  const float hr = has ? h_red[(long)n * 256 + t] : hn[(long)n * 256 + t];
  const float nv = sg * hr + (1.f - sg) * hv;
  h[(long)n * 256 + t] = nv;
  h_bf[(long)n * 256 + t] = f2bf(nv);
}

// ---------------------------------------------------------------------------
extern "C" void kernel_launch(void* const* d_in, const int* in_sizes, int n_in,
                              void* d_out, int out_size, void* d_ws, size_t ws_size,
                              hipStream_t stream)
{
  const float* node_features = (const float*)d_in[0];
  const float* edge_features = (const float*)d_in[1];
  const int*   src           = (const int*)d_in[2];
  const int*   dst           = (const int*)d_in[3];
  const float* W_node_in  = (const float*)d_in[4];
  const float* b_node_in  = (const float*)d_in[5];
  const float* W_edge_in  = (const float*)d_in[6];
  const float* b_edge_in  = (const float*)d_in[7];
  const float* W_node_out = (const float*)d_in[8];
  const float* b_node_out = (const float*)d_in[9];
  const float* node_fc_W  = (const float*)d_in[10];
  const float* node_fc_b  = (const float*)d_in[11];
  const float* edge_fc_W  = (const float*)d_in[12];
  const float* edge_fc_b  = (const float*)d_in[13];
  const float* node_att_W = (const float*)d_in[14];
  const float* rel_att_W  = (const float*)d_in[15];
  const float* rel_att_W1 = (const float*)d_in[16];
  const float* rel_att_b1 = (const float*)d_in[17];
  const float* rel_att_W2 = (const float*)d_in[18];
  const float* rel_att_b2 = (const float*)d_in[19];
  const float* out_fc_W   = (const float*)d_in[20];
  const float* out_fc_b   = (const float*)d_in[21];
  const float* gate_W     = (const float*)d_in[22];
  const float* gate_b     = (const float*)d_in[23];

  const int N = in_sizes[0] / 768;
  const int E = in_sizes[2];

  char* p = (char*)d_ws;
  auto alloc = [&](size_t bytes) -> void* {
    void* r = (void*)p; p += (bytes + 255) & ~(size_t)255; return r;
  };
  float* h      = (float*)alloc((size_t)N * 256 * 4);
  float* hn     = (float*)alloc((size_t)N * 256 * 4);
  float* a_buf  = (float*)alloc((size_t)E * 4);
  float* g_buf  = (float*)alloc((size_t)E * 3 * 4);
  float* alpha  = (float*)alloc((size_t)E * 4);
  float* beta   = (float*)alloc((size_t)E * 3 * 4);
  float* h_red  = (float*)alloc((size_t)N * 256 * 4);
  __bf16* nf_bf    = (__bf16*)alloc((size_t)N * 768 * 2);
  __bf16* ef_bf    = (__bf16*)alloc((size_t)E * 128 * 2);
  __bf16* h_bf     = (__bf16*)alloc((size_t)N * 256 * 2);
  __bf16* hn_bf    = (__bf16*)alloc((size_t)N * 256 * 2);
  __bf16* eh_bf    = (__bf16*)alloc((size_t)E * 256 * 2);
  __bf16* he_bf    = (__bf16*)alloc((size_t)E * 256 * 2);
  __bf16* agg_a_bf = (__bf16*)alloc((size_t)N * 256 * 2);
  __bf16* agg_m_bf = (__bf16*)alloc((size_t)N * 768 * 2);
  __bf16* cat_bf   = (__bf16*)alloc((size_t)N * 512 * 2);
  __bf16* Wni_t  = (__bf16*)alloc((size_t)768 * 256 * 2);
  __bf16* Wei_t  = (__bf16*)alloc((size_t)128 * 256 * 2);
  __bf16* nfc_t  = (__bf16*)alloc((size_t)2 * 65536 * 2);
  __bf16* efc_t  = (__bf16*)alloc((size_t)2 * 65536 * 2);
  __bf16* W1_t   = (__bf16*)alloc((size_t)6 * 65536 * 2);
  __bf16* Wrel_t = (__bf16*)alloc((size_t)2 * 768 * 256 * 2);
  __bf16* ofc_t  = (__bf16*)alloc((size_t)2 * 512 * 256 * 2);
  __bf16* Wout_t = (__bf16*)alloc((size_t)65536 * 2);
  __bf16* Wbar_t = (__bf16*)alloc((size_t)65536 * 2);

  const dim3 blk(256);
  const dim3 gN(2, (N + TM - 1) / TM);
  const dim3 gE(2, (E + TM - 1) / TM);
  const int T64K = 65536 / 256;

  // ---- operand prep (bf16 casts + weight transposes) ----
  cast_kernel<<<(int)(((long)N * 768 + 255) / 256), blk, 0, stream>>>(node_features, nf_bf, (long)N * 768);
  cast_kernel<<<(int)(((long)E * 128 + 255) / 256), blk, 0, stream>>>(edge_features, ef_bf, (long)E * 128);
  tcast_kernel<<<(768 * 256 + 255) / 256, blk, 0, stream>>>(W_node_in, Wni_t, 768, 256);
  tcast_kernel<<<(128 * 256 + 255) / 256, blk, 0, stream>>>(W_edge_in, Wei_t, 128, 256);
  tcast_kernel<<<T64K, blk, 0, stream>>>(W_node_out, Wout_t, 256, 256);
  for (int l = 0; l < 2; ++l) {
    tcast_kernel<<<T64K, blk, 0, stream>>>(node_fc_W + (size_t)l * 65536, nfc_t + (size_t)l * 65536, 256, 256);
    tcast_kernel<<<T64K, blk, 0, stream>>>(edge_fc_W + (size_t)l * 65536, efc_t + (size_t)l * 65536, 256, 256);
    for (int m = 0; m < 3; ++m)
      tcast_kernel<<<T64K, blk, 0, stream>>>(rel_att_W1 + (size_t)(l * 3 + m) * 65536,
                                             W1_t + (size_t)(l * 3 + m) * 65536, 256, 256);
    tcast_kernel<<<(768 * 256 + 255) / 256, blk, 0, stream>>>(rel_att_W + (size_t)l * 3 * 65536,
                                                              Wrel_t + (size_t)l * 768 * 256, 768, 256);
    tcast_kernel<<<(512 * 256 + 255) / 256, blk, 0, stream>>>(out_fc_W + (size_t)l * 131072,
                                                              ofc_t + (size_t)l * 512 * 256, 512, 256);
  }

  // ---- input projections (leaky relu) ----
  gemm_abf16<1><<<gN, blk, 0, stream>>>(nf_bf, Wni_t, b_node_in, h, h_bf, N, 768, 256, 0, 1.f);
  gemm_abf16<1><<<gE, blk, 0, stream>>>(ef_bf, Wei_t, b_edge_in, nullptr, eh_bf, E, 128, 256, 0, 1.f);

  for (int l = 0; l < 2; ++l) {
    gemm_abf16<0><<<gN, blk, 0, stream>>>(h_bf, nfc_t + (size_t)l * 65536, node_fc_b + l * 256,
                                          hn, hn_bf, N, 256, 256, 0, 1.f);
    gemm_abf16<0><<<gE, blk, 0, stream>>>(eh_bf, efc_t + (size_t)l * 65536, edge_fc_b + l * 256,
                                          nullptr, he_bf, E, 256, 256, 0, 1.f);

    edge_dot_kernel<<<(E + 7) / 8, blk, 0, stream>>>(hn_bf, src, dst, a_buf, E);

    rel_att_kernel<<<dim3((E + TM - 1) / TM, 3), blk, 0, stream>>>(
        he_bf, W1_t + (size_t)l * 3 * 65536, rel_att_b1 + l * 768,
        rel_att_W2 + l * 768, rel_att_b2 + l * 3, g_buf, E);

    seg_softmax_kernel<<<(N + 7) / 8, blk, 0, stream>>>(a_buf, g_buf, dst, alpha, beta, N, E);

    agg_kernel<<<N, blk, 0, stream>>>(hn_bf, src, dst, alpha, beta, agg_a_bf, agg_m_bf, E);

    wbar_t_kernel<<<T64K, blk, 0, stream>>>(node_att_W + (size_t)l * 3 * 65536, Wbar_t);

    gemm_abf16<0><<<gN, blk, 0, stream>>>(agg_a_bf, Wbar_t, nullptr, nullptr, cat_bf, N, 256, 512, 0, 1.f);
    gemm_abf16<0><<<gN, blk, 0, stream>>>(agg_m_bf, Wrel_t + (size_t)l * 768 * 256, nullptr, nullptr,
                                          cat_bf, N, 768, 512, 256, 1.f / 3.f);

    gemm_abf16<2><<<gN, blk, 0, stream>>>(cat_bf, ofc_t + (size_t)l * 512 * 256, out_fc_b + l * 256,
                                          h_red, nullptr, N, 512, 256, 0, 1.f);

    gate_update_kernel<<<N, blk, 0, stream>>>(h, h_bf, h_red, hn, dst, gate_W + l * 256, gate_b + l, E);
  }

  gemm_abf16<1><<<gN, blk, 0, stream>>>(h_bf, Wout_t, b_node_out, (float*)d_out, nullptr, N, 256, 256, 0, 1.f);
}
